// MultiheadAttention_with_jax_24180665877125
// MI455X (gfx1250) — matrix operand order repaired and checked by reading, not compiled, not run on hardware
//
#include <hip/hip_runtime.h>
#include <hip/hip_bf16.h>

typedef __attribute__((ext_vector_type(16))) __bf16 v16bf;
typedef __attribute__((ext_vector_type(8)))  __bf16 bf16x8;
typedef __attribute__((ext_vector_type(4)))  __bf16 bf16x4;
typedef __attribute__((ext_vector_type(8)))  float  v8f;
typedef __attribute__((ext_vector_type(4)))  float  f32x4;

// Problem constants
constexpr int kB = 4, kT = 4096, kE = 2048, kH = 16, kD = 128, kBH = 64;

// ---------------------------------------------------------------------------
// WMMA helpers (CDNA5: v_wmma_f32_16x16x32_bf16, wave32)
// ---------------------------------------------------------------------------
__device__ inline v8f wmma_bf16(v16bf a, v16bf b, v8f c) {
  // (neg_a, A, neg_b, B, c_mod, C, reuse_a, reuse_b)
  return __builtin_amdgcn_wmma_f32_16x16x32_bf16(false, a, false, b,
                                                 (short)0, c, false, false);
}

// A-matrix 16x32 bf16 fragment (ISA 7.12.2): lane L holds M = L%16.
// Lanes 0-15: K = {0..7, 16..23}; lanes 16-31: K = {8..15, 24..31}.
__device__ inline v16bf frag_a(const __bf16* base, int row, int ldr, int kofs,
                               int lane) {
  const __bf16* p = base + (row + (lane & 15)) * ldr + kofs + ((lane >> 4) << 3);
  bf16x8 lo = *(const bf16x8*)p;          // K = kh .. kh+7
  bf16x8 hi = *(const bf16x8*)(p + 16);   // K = kh+16 .. kh+23
  v16bf r;
#pragma unroll
  for (int i = 0; i < 8; ++i) { r[i] = lo[i]; r[i + 8] = hi[i]; }
  return r;
}

// B-matrix 32x16 bf16 fragment: lane L holds N = L%16.
// Lanes 0-15: K = {0..7, 16..23}; lanes 16-31: K = {8..15, 24..31}. Source buffer is stored
// N-major (row = n, contiguous in k), so each lane reads two 16-byte chunks.
__device__ inline v16bf frag_b(const __bf16* base, int col, int ldr, int kofs,
                               int lane) {
  const __bf16* p = base + (col + (lane & 15)) * ldr + kofs + ((lane >> 4) << 3);
  bf16x8 lo = *(const bf16x8*)p;          // K = kb .. kb+7
  bf16x8 hi = *(const bf16x8*)(p + 16);   // K = kb+16 .. kb+23
  v16bf r;
#pragma unroll
  for (int i = 0; i < 8; ++i) { r[i] = lo[i]; r[i + 8] = hi[i]; }
  return r;
}

// ---------------------------------------------------------------------------
// Kernel 0: fp32 -> bf16 weight conversion (layout preserved: W is (n,k)
// row-major, which is exactly the N-major B-operand layout we want).
// ---------------------------------------------------------------------------
__global__ __launch_bounds__(256) void cvt_f32_bf16(
    const float* __restrict__ src, __bf16* __restrict__ dst, int n4) {
  int i = blockIdx.x * 256 + threadIdx.x;
  if (i < n4) {
    f32x4 v = *(const f32x4*)(src + (size_t)i * 4);
    bf16x4 o;
    o[0] = (__bf16)v[0]; o[1] = (__bf16)v[1];
    o[2] = (__bf16)v[2]; o[3] = (__bf16)v[3];
    *(bf16x4*)(dst + (size_t)i * 4) = o;
  }
}

// ---------------------------------------------------------------------------
// GEMM: C[M=16384, N=2048] = A[M,2048] * W^T + bias  (W given (n,k) bf16).
// 128x128 tile per block, 8 waves, each wave 32(M)x64(N) = 8 WMMA tiles.
// Double-buffered LDS (register-staged prefetch of tile k+1 while computing
// tile k) + global_prefetch of tile k+2.
// MODE 0: A fp32 (input activations); out = bf16 split-head [t][bh][d] * scale
// MODE 1: A bf16 (attention output);  out = fp32 row-major d_out
// ---------------------------------------------------------------------------
template <int MODE>
__global__ __launch_bounds__(256) void gemm128(
    const void* __restrict__ Ap, const __bf16* __restrict__ Bw,
    const float* __restrict__ bias, float scale, void* __restrict__ outp) {
  constexpr int LD = 40;  // 80-byte rows: conflict-free, 16B aligned
  __shared__ __bf16 As[2][128 * LD];
  __shared__ __bf16 Bs[2][128 * LD];

  const int tid  = threadIdx.x;
  const int lane = tid & 31;
  const int wid  = tid >> 5;        // 8 waves (wave32)
  const int wm   = wid & 3;         // M sub-block (32 rows)
  const int wn   = wid >> 2;        // N sub-block (64 cols)
  const int arow0 = blockIdx.y * 128;
  const int brow0 = blockIdx.x * 128;

  v8f acc[2][4] = {};

  f32x4  a4[4];   // MODE 0 staging (fp32 A)
  bf16x8 a8[2];   // MODE 1 staging (bf16 A)
  bf16x8 b8[2];   // B staging

  auto load_global = [&](int k0) {
    if constexpr (MODE == 0) {
      const float* A = (const float*)Ap;
#pragma unroll
      for (int i = 0; i < 4; ++i) {
        int idx = tid + i * 256;          // 1024 float4 chunks
        int r = idx >> 3, c = idx & 7;
        a4[i] = *(const f32x4*)(A + (size_t)(arow0 + r) * kE + k0 + c * 4);
      }
    } else {
      const __bf16* A = (const __bf16*)Ap;
#pragma unroll
      for (int i = 0; i < 2; ++i) {
        int idx = tid + i * 256;          // 512 chunks of 8 bf16
        int r = idx >> 2, c = idx & 3;
        a8[i] = *(const bf16x8*)(A + (size_t)(arow0 + r) * kE + k0 + c * 8);
      }
    }
#pragma unroll
    for (int i = 0; i < 2; ++i) {
      int idx = tid + i * 256;
      int r = idx >> 2, c = idx & 3;
      b8[i] = *(const bf16x8*)(Bw + (size_t)(brow0 + r) * kE + k0 + c * 8);
    }
  };

  auto store_lds = [&](int buf) {
    if constexpr (MODE == 0) {
#pragma unroll
      for (int i = 0; i < 4; ++i) {
        int idx = tid + i * 256;
        int r = idx >> 3, c = idx & 7;
        bf16x4 pv;
        pv[0] = (__bf16)a4[i][0]; pv[1] = (__bf16)a4[i][1];
        pv[2] = (__bf16)a4[i][2]; pv[3] = (__bf16)a4[i][3];
        *(bf16x4*)&As[buf][r * LD + c * 4] = pv;
      }
    } else {
#pragma unroll
      for (int i = 0; i < 2; ++i) {
        int idx = tid + i * 256;
        int r = idx >> 2, c = idx & 3;
        *(bf16x8*)&As[buf][r * LD + c * 8] = a8[i];
      }
    }
#pragma unroll
    for (int i = 0; i < 2; ++i) {
      int idx = tid + i * 256;
      int r = idx >> 2, c = idx & 3;
      *(bf16x8*)&Bs[buf][r * LD + c * 8] = b8[i];
    }
  };

  auto prefetch_tile = [&](int k0) {   // global_prefetch_b8 (gfx1250)
    if constexpr (MODE == 0) {
      const float* A = (const float*)Ap;
      int r = tid >> 3, c = tid & 7;
      __builtin_prefetch(A + (size_t)(arow0 + r) * kE + k0 + c * 4, 0, 3);
      __builtin_prefetch(A + (size_t)(arow0 + 32 + r) * kE + k0 + c * 4, 0, 3);
      __builtin_prefetch(A + (size_t)(arow0 + 64 + r) * kE + k0 + c * 4, 0, 3);
      __builtin_prefetch(A + (size_t)(arow0 + 96 + r) * kE + k0 + c * 4, 0, 3);
    } else {
      const __bf16* A = (const __bf16*)Ap;
      int r = tid >> 1, c = tid & 1;
      __builtin_prefetch(A + (size_t)(arow0 + r) * kE + k0 + c * 16, 0, 3);
    }
    int r = tid >> 1, c = tid & 1;
    __builtin_prefetch(Bw + (size_t)(brow0 + r) * kE + k0 + c * 16, 0, 3);
  };

  load_global(0);
  store_lds(0);
  __syncthreads();

  int cur = 0;
  for (int k0 = 0; k0 < kE; k0 += 32) {
    const bool has_next = (k0 + 32) < kE;
    if (has_next) {
      load_global(k0 + 32);             // stage next tile in registers
      if (k0 + 64 < kE) prefetch_tile(k0 + 64);
    }

    // Compute current tile from LDS[cur]
    v16bf a0 = frag_a(&As[cur][0], wm * 32, LD, 0, lane);
    v16bf a1 = frag_a(&As[cur][0], wm * 32 + 16, LD, 0, lane);
#pragma unroll
    for (int j = 0; j < 4; ++j) {
      v16bf b = frag_b(&Bs[cur][0], wn * 64 + j * 16, LD, 0, lane);
      acc[0][j] = wmma_bf16(a0, b, acc[0][j]);
      acc[1][j] = wmma_bf16(a1, b, acc[1][j]);
    }

    if (has_next) store_lds(cur ^ 1);   // fill the other buffer
    __syncthreads();
    cur ^= 1;
  }

  // Epilogue. C layout: lane holds N = lane%16; VGPR v -> M = v (+8 if lane>=16)
#pragma unroll
  for (int i = 0; i < 2; ++i) {
#pragma unroll
    for (int j = 0; j < 4; ++j) {
      int mbase = arow0 + wm * 32 + i * 16 + ((lane >> 4) << 3);
      int n = brow0 + wn * 64 + j * 16 + (lane & 15);
      float bv = bias[n];
#pragma unroll
      for (int v = 0; v < 8; ++v) {
        int m = mbase + v;
        float val = (acc[i][j][v] + bv) * scale;
        if constexpr (MODE == 0) {
          // split heads: m = b*T + t ; n = head*128 + d -> [t][b*16+head][d]
          int b = m >> 12, t = m & 4095;
          int head = n >> 7, d = n & 127;
          ((__bf16*)outp)[((size_t)t * kBH + (b * kH + head)) * kD + d] =
              (__bf16)val;
        } else {
          ((float*)outp)[(size_t)m * kE + n] = val;
        }
      }
    }
  }
}

// ---------------------------------------------------------------------------
// Attention: one block per t. S = Q*K^T (64x64, contraction over D=128),
// row softmax over key axis, O = S*V (64x128). All matmuls via WMMA bf16.
// ---------------------------------------------------------------------------
__global__ __launch_bounds__(256) void attn_kernel(
    const __bf16* __restrict__ qs, const __bf16* __restrict__ ks,
    const __bf16* __restrict__ vs, __bf16* __restrict__ attn) {
  __shared__ __align__(16) char smem[53248];
  __bf16* Qs = (__bf16*)smem;            // 64 x 136 rows (272B, 16B aligned)
  __bf16* Ks = (__bf16*)(smem + 17408);  // 64 x 136
  __bf16* Vt = (__bf16*)(smem + 34816);  // 128 x 72 (V transposed: row = d)
  float*  Sf = (float*)smem;             // reuse Qs region: 64 x 68 fp32
  __bf16* Wb = (__bf16*)(smem + 17408);  // reuse Ks region: 64 x 72 bf16

  const int t    = blockIdx.x;
  const int tid  = threadIdx.x;
  const int lane = tid & 31;
  const int wid  = tid >> 5;
  const size_t base = (size_t)t * (kBH * kD);

  // Load Q,K (row-major, padded) and V transposed.
#pragma unroll
  for (int i = 0; i < 4; ++i) {
    int idx = tid + i * 256;             // 1024 chunks of 8 bf16
    int r = idx >> 4, c = idx & 15;
    bf16x8 vq = *(const bf16x8*)(qs + base + r * kD + c * 8);
    bf16x8 vk = *(const bf16x8*)(ks + base + r * kD + c * 8);
    bf16x8 vv = *(const bf16x8*)(vs + base + r * kD + c * 8);
    *(bf16x8*)&Qs[r * 136 + c * 8] = vq;
    *(bf16x8*)&Ks[r * 136 + c * 8] = vk;
#pragma unroll
    for (int j = 0; j < 8; ++j) Vt[(c * 8 + j) * 72 + r] = vv[j];
  }
  __syncthreads();

  // S = Q*K^T : 4x4 tiles over 8 waves -> each wave 1 M-tile x 2 N-tiles
  const int mt = (wid & 3) * 16;
  const int nt = (wid >> 2) * 32;
  v8f s0 = {}, s1 = {};
#pragma unroll
  for (int s = 0; s < 4; ++s) {          // D = 128 in 4 K-steps
    v16bf aq  = frag_a(Qs, mt, 136, s * 32, lane);
    v16bf bk0 = frag_b(Ks, nt, 136, s * 32, lane);
    v16bf bk1 = frag_b(Ks, nt + 16, 136, s * 32, lane);
    s0 = wmma_bf16(aq, bk0, s0);
    s1 = wmma_bf16(aq, bk1, s1);
  }
  __syncthreads();  // all waves done reading Qs before Sf (aliased) is written
  {
    int mb = mt + ((lane >> 4) << 3);
    int n0 = nt + (lane & 15);
#pragma unroll
    for (int v = 0; v < 8; ++v) {
      Sf[(mb + v) * 68 + n0]      = s0[v];
      Sf[(mb + v) * 68 + n0 + 16] = s1[v];
    }
  }
  __syncthreads();

  // Row softmax (64 rows of 64), scaling already folded into q projection.
  if (tid < 64) {
    float* r = Sf + tid * 68;
    float mx = r[0];
#pragma unroll 8
    for (int j = 1; j < 64; ++j) mx = fmaxf(mx, r[j]);
    float sum = 0.f;
#pragma unroll 8
    for (int j = 0; j < 64; ++j) { float e = __expf(r[j] - mx); sum += e; r[j] = e; }
    float inv = 1.f / sum;
#pragma unroll 8
    for (int j = 0; j < 64; ++j) Wb[tid * 72 + j] = (__bf16)(r[j] * inv);
  }
  __syncthreads();

  // O = W(64x64) * V(64x128): 4x8 tiles over 8 waves -> 1 M-tile x 4 N-tiles
  v8f o[4] = {};
  const int nb = (wid >> 2) * 64;
#pragma unroll
  for (int s = 0; s < 2; ++s) {          // K = 64 in 2 steps
    v16bf aw = frag_a(Wb, mt, 72, s * 32, lane);
#pragma unroll
    for (int j = 0; j < 4; ++j) {
      v16bf bv = frag_b(Vt, nb + j * 16, 72, s * 32, lane);
      o[j] = wmma_bf16(aw, bv, o[j]);
    }
  }
  // Write back merged-head layout: row = b*T + t, col = head*128 + d
  {
    int mb = mt + ((lane >> 4) << 3);
#pragma unroll
    for (int j = 0; j < 4; ++j) {
      int n = nb + j * 16 + (lane & 15);   // d
#pragma unroll
      for (int v = 0; v < 8; ++v) {
        int m = mb + v;                     // query bh index
        int b = m >> 4, head = m & 15;
        attn[((size_t)(b * kT + t)) * kE + head * kD + n] = (__bf16)o[j][v];
      }
    }
  }
}

// ---------------------------------------------------------------------------
extern "C" void kernel_launch(void* const* d_in, const int* in_sizes, int n_in,
                              void* d_out, int out_size, void* d_ws,
                              size_t ws_size, hipStream_t stream) {
  const float* query = (const float*)d_in[0];
  const float* key   = (const float*)d_in[1];
  const float* value = (const float*)d_in[2];
  const float* Wq = (const float*)d_in[3];
  const float* bq = (const float*)d_in[4];
  const float* Wk = (const float*)d_in[5];
  const float* bk = (const float*)d_in[6];
  const float* Wv = (const float*)d_in[7];
  const float* bv = (const float*)d_in[8];
  const float* Wo = (const float*)d_in[9];
  const float* bo = (const float*)d_in[10];

  char* ws = (char*)d_ws;
  __bf16* wq_b = (__bf16*)(ws);
  __bf16* wk_b = (__bf16*)(ws + 8388608);
  __bf16* wv_b = (__bf16*)(ws + 16777216);
  __bf16* wo_b = (__bf16*)(ws + 25165824);
  __bf16* qsb  = (__bf16*)(ws + 33554432);   // [T][BH][D] bf16
  __bf16* ksb  = (__bf16*)(ws + 100663296);
  __bf16* vsb  = (__bf16*)(ws + 167772160);
  __bf16* attn = (__bf16*)(ws + 234881024);  // [B*T][E] bf16

  const int n4 = kE * kE / 4;  // 1,048,576
  cvt_f32_bf16<<<4096, 256, 0, stream>>>(Wq, wq_b, n4);
  cvt_f32_bf16<<<4096, 256, 0, stream>>>(Wk, wk_b, n4);
  cvt_f32_bf16<<<4096, 256, 0, stream>>>(Wv, wv_b, n4);
  cvt_f32_bf16<<<4096, 256, 0, stream>>>(Wo, wo_b, n4);

  dim3 gg(kE / 128, (kB * kT) / 128);  // (16, 128)
  // q gets both scalings folded: D^-0.5 * D^-0.5 = 1/128
  gemm128<0><<<gg, 256, 0, stream>>>(query, wq_b, bq, 1.0f / 128.0f, qsb);
  gemm128<0><<<gg, 256, 0, stream>>>(key,   wk_b, bk, 1.0f, ksb);
  gemm128<0><<<gg, 256, 0, stream>>>(value, wv_b, bv, 1.0f, vsb);

  attn_kernel<<<kT, 256, 0, stream>>>(qsb, ksb, vsb, attn);

  gemm128<1><<<gg, 256, 0, stream>>>(attn, wo_b, bo, 1.0f, d_out);
}
